// Decoder_43173011259949
// MI455X (gfx1250) — matrix operand order repaired and checked by reading, not compiled, not run on hardware
//
#include <hip/hip_runtime.h>

// ---------------------------------------------------------------------------
// Transformer decoder forward for MI455X (gfx1250), bf16 WMMA everywhere.
//
// Roofline: logits write (524MB) + weight reads (~200MB) dominate at
// 23.3 TB/s (~35us floor); 334 GFLOP of bf16 WMMA is far below the matrix
// ceiling, so everything minimizes DRAM traffic and hides latency:
//  - flash-attention (no S x S score materialization)
//  - bf16 LDS staging via packed v_cvt_pk_bf16_f32 (1 VALU per pair,
//    b32 LDS stores, no extra DRAM passes)
//  - double-buffered LDS: global loads of tile k+1 overlap WMMA on tile k,
//    one barrier per K-step
//  - global_prefetch_b8 hints two tiles ahead
//  - branchless fused bias/ReLU/BN epilogues.
// ---------------------------------------------------------------------------

#define D_MODEL 512
#define NHEAD   8
#define DHEAD   64
#define NLAYER  4
#define SEQ     1024
#define SEQ_ENC 1024
#define BATCH   4
#define VOCAB   32000
#define DFF     2048
#define BN_EPS  1e-3f
#define NEG_BIG -3.402823466e38f

typedef __bf16 bhalf;
typedef bhalf          v16bf __attribute__((ext_vector_type(16)));
typedef float          v8f   __attribute__((ext_vector_type(8)));
typedef unsigned short ushort_t;
typedef ushort_t       v8us  __attribute__((ext_vector_type(8)));

// packed RNE f32x2 -> bf16x2 (low = a, high = b) in one VALU op
__device__ __forceinline__ unsigned pk_bf16(float a, float b) {
  unsigned r;
  asm("v_cvt_pk_bf16_f32 %0, %1, %2" : "=v"(r) : "v"(a), "v"(b));
  return r;
}

// scalar RNE f32 -> bf16 bit pattern (used only for the P scatter)
__device__ __forceinline__ ushort_t f2bf(float f) {
  unsigned int u = __builtin_bit_cast(unsigned int, f);
  u += 0x7FFFu + ((u >> 16) & 1u);
  return (ushort_t)(u >> 16);
}

// assemble a 16-element bf16 fragment from two 16B-aligned LDS chunks
__device__ __forceinline__ v16bf frag16(const ushort_t* p0, const ushort_t* p1) {
  union { v8us u[2]; v16bf bf; } x;
  x.u[0] = *(const v8us*)p0;
  x.u[1] = *(const v8us*)p1;
  return x.bf;
}

__device__ __forceinline__ v8f zero8() {
  v8f z;
#pragma unroll
  for (int i = 0; i < 8; ++i) z[i] = 0.0f;
  return z;
}

// ---------------------------------------------------------------------------
// GEMM tile staging helpers (64x32 A tile, 32x64 B tile, 128 threads).
// All data handled as bf16 pairs: A pairs along K (contiguous, -> b64 loads),
// B pairs across two K-rows (both reads N-coalesced).
// ---------------------------------------------------------------------------
__device__ __forceinline__ void stage_gload(const float* __restrict__ A,
                                            const float* __restrict__ B,
                                            int m0, int n0, int k0, int N,
                                            int K, int tid, float* ar,
                                            float* br) {
#pragma unroll
  for (int i = 0; i < 8; ++i) {            // 1024 A-pairs (64 x 16)
    int idx = tid + 128 * i;
    int r = idx >> 4, c2 = idx & 15;
    const float* p = &A[(size_t)(m0 + r) * K + k0 + 2 * c2];
    ar[2 * i] = p[0];
    ar[2 * i + 1] = p[1];
  }
#pragma unroll
  for (int i = 0; i < 8; ++i) {            // 1024 B-pairs (16 x 64)
    int idx = tid + 128 * i;
    int n = idx & 63, kp = idx >> 6;
    const float* p = &B[(size_t)(k0 + 2 * kp) * N + n0 + n];
    br[2 * i] = p[0];
    br[2 * i + 1] = p[(size_t)N];
  }
}

__device__ __forceinline__ void stage_sstore(ushort_t (*__restrict__ As)[40],
                                             ushort_t (*__restrict__ BsT)[40],
                                             int tid, const float* ar,
                                             const float* br) {
#pragma unroll
  for (int i = 0; i < 8; ++i) {
    int idx = tid + 128 * i;
    int r = idx >> 4, c2 = idx & 15;
    *(unsigned*)&As[r][2 * c2] = pk_bf16(ar[2 * i], ar[2 * i + 1]);
  }
#pragma unroll
  for (int i = 0; i < 8; ++i) {
    int idx = tid + 128 * i;
    int n = idx & 63, kp = idx >> 6;       // BsT[n][k] pairs along k
    *(unsigned*)&BsT[n][2 * kp] = pk_bf16(br[2 * i], br[2 * i + 1]);
  }
}

// ---------------------------------------------------------------------------
// Generic GEMM: C[M,N] = act(A[M,K] @ B[K,N] + bias), fp32 in/out, bf16 WMMA.
// Tile 64x64, block = 128 threads = 4 waves, each wave 32x32 (2x2 WMMA accs).
// Double-buffered LDS, one barrier per K-step.
// Requires M%64==0, N%64==0, K%32==0 (true for all GEMMs in this model).
// ---------------------------------------------------------------------------
__global__ __launch_bounds__(128)
void gemm_bf16_kernel(const float* __restrict__ A, const float* __restrict__ B,
                      const float* __restrict__ bias, float* __restrict__ C,
                      int M, int N, int K, int relu) {
  (void)M;
  // 80B row stride keeps every fragment chunk 16B aligned
  __shared__ __align__(16) ushort_t As[2][64][40];   // [buf][m][k] bf16
  __shared__ __align__(16) ushort_t BsT[2][64][40];  // [buf][n][k] bf16

  const int tid = threadIdx.x;
  const int lane = tid & 31;
  const int w  = tid >> 5;
  const int wm = w >> 1, wn = w & 1;
  const int m0 = blockIdx.y * 64;
  const int n0 = blockIdx.x * 64;
  const int hi = lane >> 4;     // lane half
  const int lo = lane & 15;
  const int kb = hi * 8;        // A-layout K chunk base
  const float relu_floor = relu ? 0.0f : NEG_BIG;

  v8f acc[2][2];
  acc[0][0] = zero8(); acc[0][1] = zero8();
  acc[1][0] = zero8(); acc[1][1] = zero8();

  float ar[16], br[16];
  stage_gload(A, B, m0, n0, 0, N, K, tid, ar, br);
  stage_sstore(As[0], BsT[0], tid, ar, br);
  __syncthreads();

  int cur = 0;
  for (int k0 = 0; k0 < K; k0 += 32) {
    const int knext = k0 + 32;
    const bool has_next = knext < K;
    if (has_next) {
      // overlap: fetch tile k+1 into registers while WMMAs run on tile k
      stage_gload(A, B, m0, n0, knext, N, K, tid, ar, br);
      // hint tile k+2 toward the caches
      const int knn = k0 + 64;
      if (knn < K) {
        __builtin_prefetch(A + (size_t)(m0 + (tid & 63)) * K + knn +
                               ((tid >> 6) << 4), 0, 0);
        __builtin_prefetch(B + (size_t)(knn + (tid & 31)) * N + n0 +
                               ((tid >> 5) << 4), 0, 0);
      }
    }

    v16bf af[2], bfr[2];
#pragma unroll
    for (int r = 0; r < 2; ++r) {
      int row = wm * 32 + r * 16 + lo;
      // A 16x32 layout: elems 0-7 = K kb..kb+7, elems 8-15 = K kb+16..kb+23
      af[r] = frag16(&As[cur][row][kb], &As[cur][row][kb + 16]);
    }
#pragma unroll
    for (int c = 0; c < 2; ++c) {
      int col = wn * 32 + c * 16 + lo;
      // B 32x16 layout: elems 0-7 = K kb..kb+7, elems 8-15 = K kb+16..kb+23
      bfr[c] = frag16(&BsT[cur][col][kb], &BsT[cur][col][kb + 16]);
    }
#pragma unroll
    for (int r = 0; r < 2; ++r)
#pragma unroll
      for (int c = 0; c < 2; ++c)
        acc[r][c] = __builtin_amdgcn_wmma_f32_16x16x32_bf16(
            false, af[r], false, bfr[c], (short)0, acc[r][c], false, false);

    if (has_next) {
      // write buffer nobody reads; single barrier publishes it
      stage_sstore(As[1 - cur], BsT[1 - cur], tid, ar, br);
      __syncthreads();
      cur ^= 1;
    }
  }

  // epilogue: bias + branchless ReLU, store. C layout: elem j -> M=j+8*hi, N=lo
#pragma unroll
  for (int r = 0; r < 2; ++r) {
#pragma unroll
    for (int c = 0; c < 2; ++c) {
      int col = n0 + wn * 32 + c * 16 + lo;
      float bv = bias[col];
#pragma unroll
      for (int j = 0; j < 8; ++j) {
        int row = m0 + wm * 32 + r * 16 + j + 8 * hi;
        C[(size_t)row * N + col] = fmaxf(acc[r][c][j] + bv, relu_floor);
      }
    }
  }
}

// ---------------------------------------------------------------------------
// Flash attention: grid (Sq/64, H, B), block = 128 threads = 4 waves.
// Each wave owns 16 q-rows; K/V streamed in 32-key bf16 LDS tiles.
// causal==1 -> strictly-past mask (key < query); fully-masked rows emit 0.
// ---------------------------------------------------------------------------
__global__ __launch_bounds__(128)
void attn_kernel(const float* __restrict__ Q, const float* __restrict__ K,
                 const float* __restrict__ V, float* __restrict__ O,
                 int Sq, int Sk, int causal) {
  __shared__ __align__(16) ushort_t Ks[32][72];      // [key][d]   (144B rows)
  __shared__ __align__(16) ushort_t VsT[64][40];     // [d][key]   (80B rows)
  __shared__ __align__(16) ushort_t Ps[4][16][40];   // per-wave P tile

  const int tid = threadIdx.x;
  const int lane = tid & 31;
  const int w  = tid >> 5;
  const int bx = blockIdx.x;    // q tile
  const int h  = blockIdx.y;
  const int b  = blockIdx.z;
  const int hi = lane >> 4, lo = lane & 15, kb = hi * 8;
  const int q0w = bx * 64 + w * 16;   // first q row of this wave

  // resident Q fragments (16 rows x 64, pre-scaled by 1/sqrt(DH)=0.125)
  v16bf qf[2];
  {
    const float* qrow = Q + (size_t)(b * Sq + q0w + lo) * D_MODEL + h * DHEAD;
#pragma unroll
    for (int t = 0; t < 2; ++t) {
      union { unsigned u[8]; v16bf bf; } x;
#pragma unroll
      for (int i = 0; i < 4; ++i) {
        x.u[i]     = pk_bf16(qrow[t * 32 + kb + 2 * i] * 0.125f,
                             qrow[t * 32 + kb + 2 * i + 1] * 0.125f);
        x.u[4 + i] = pk_bf16(qrow[t * 32 + kb + 16 + 2 * i] * 0.125f,
                             qrow[t * 32 + kb + 17 + 2 * i] * 0.125f);
      }
      qf[t] = x.bf;
    }
  }

  v8f oacc[4];
#pragma unroll
  for (int c = 0; c < 4; ++c) oacc[c] = zero8();
  float mst[8], lst[8];
#pragma unroll
  for (int j = 0; j < 8; ++j) { mst[j] = -1e30f; lst[j] = 0.0f; }

  const int ktend = causal ? (bx * 64 + 64) : Sk;  // strictly-past keys only
  for (int kt = 0; kt < ktend; kt += 32) {
    // prefetch next K/V tile while this one is consumed
    {
      int ktn = kt + 32;
      if (ktn < ktend) {
        size_t g = (size_t)(b * Sk + ktn + (tid >> 2)) * D_MODEL + h * DHEAD +
                   ((tid & 3) << 4);
        __builtin_prefetch(K + g, 0, 0);
        __builtin_prefetch(V + g, 0, 0);
      }
    }
    // stage K (row-major, pairs along d) and V (transposed, pairs along key)
#pragma unroll
    for (int i = 0; i < 8; ++i) {          // 1024 K-pairs
      int idx = tid + 128 * i;
      int d2 = idx & 31, kk = idx >> 5;
      size_t g = (size_t)(b * Sk + kt + kk) * D_MODEL + h * DHEAD + 2 * d2;
      *(unsigned*)&Ks[kk][2 * d2] = pk_bf16(K[g], K[g + 1]);
    }
#pragma unroll
    for (int i = 0; i < 8; ++i) {          // 1024 V-pairs
      int idx = tid + 128 * i;
      int d = idx & 63, kp = idx >> 6;
      size_t g = (size_t)(b * Sk + kt + 2 * kp) * D_MODEL + h * DHEAD + d;
      *(unsigned*)&VsT[d][2 * kp] = pk_bf16(V[g], V[g + D_MODEL]);
    }
    __syncthreads();

    // scores: 16q x 32k as two 16x16 WMMA tiles, K-dim = DH = 64 (2 WMMAs)
    v8f s[2];
    s[0] = zero8(); s[1] = zero8();
#pragma unroll
    for (int n0i = 0; n0i < 2; ++n0i) {
#pragma unroll
      for (int t = 0; t < 2; ++t) {
        v16bf kf = frag16(&Ks[n0i * 16 + lo][t * 32 + kb],
                          &Ks[n0i * 16 + lo][t * 32 + kb + 16]);
        s[n0i] = __builtin_amdgcn_wmma_f32_16x16x32_bf16(
            false, qf[t], false, kf, (short)0, s[n0i], false, false);
      }
    }

    // online softmax per accumulator row (M = j + 8*hi, N = lo)
#pragma unroll
    for (int j = 0; j < 8; ++j) {
      float v0 = s[0][j], v1 = s[1][j];
      int qr = q0w + j + 8 * hi;
      if (causal) {
        if (kt + lo >= qr)      v0 = -1e30f;
        if (kt + 16 + lo >= qr) v1 = -1e30f;
      }
      float rm = fmaxf(v0, v1);
      rm = fmaxf(rm, __shfl_xor(rm, 1));
      rm = fmaxf(rm, __shfl_xor(rm, 2));
      rm = fmaxf(rm, __shfl_xor(rm, 4));
      rm = fmaxf(rm, __shfl_xor(rm, 8));   // 16-lane halves stay separate
      float nm = fmaxf(mst[j], rm);
      float corr = __expf(mst[j] - nm);
      float p0 = (v0 < -1e29f) ? 0.0f : __expf(v0 - nm);
      float p1 = (v1 < -1e29f) ? 0.0f : __expf(v1 - nm);
      float rs = p0 + p1;
      rs += __shfl_xor(rs, 1);
      rs += __shfl_xor(rs, 2);
      rs += __shfl_xor(rs, 4);
      rs += __shfl_xor(rs, 8);
      lst[j] = lst[j] * corr + rs;
      mst[j] = nm;
#pragma unroll
      for (int c = 0; c < 4; ++c) oacc[c][j] *= corr;
      // C-layout -> A-layout reshape via per-wave LDS (same-wave LDS in-order)
      Ps[w][j + 8 * hi][lo]      = f2bf(p0);
      Ps[w][j + 8 * hi][16 + lo] = f2bf(p1);
    }

    // O += P(16x32) @ V(32x64) : 4 WMMAs over DH chunks
    v16bf pf = frag16(&Ps[w][lo][kb], &Ps[w][lo][kb + 16]);
#pragma unroll
    for (int c = 0; c < 4; ++c) {
      v16bf vf = frag16(&VsT[c * 16 + lo][kb],
                        &VsT[c * 16 + lo][kb + 16]);
      oacc[c] = __builtin_amdgcn_wmma_f32_16x16x32_bf16(
          false, pf, false, vf, (short)0, oacc[c], false, false);
    }
    __syncthreads();
  }

  // normalize (fully-masked rows -> exact zeros, matching softmax*mask) + store
#pragma unroll
  for (int j = 0; j < 8; ++j) {
    float inv = lst[j] > 0.0f ? 1.0f / lst[j] : 0.0f;
    int qr = q0w + j + 8 * hi;
#pragma unroll
    for (int c = 0; c < 4; ++c) {
      O[(size_t)(b * Sq + qr) * D_MODEL + h * DHEAD + c * 16 + lo] =
          oacc[c][j] * inv;
    }
  }
}

// ---------------------------------------------------------------------------
// Elementwise kernels
// ---------------------------------------------------------------------------
__global__ void embed_kernel(const int* __restrict__ seq,
                             const float* __restrict__ emb,
                             const float* __restrict__ pes,
                             float* __restrict__ x, int n) {
  int idx = blockIdx.x * blockDim.x + threadIdx.x;
  if (idx >= n) return;
  int bs = idx >> 9;           // / D_MODEL (512)
  int d = idx & 511;
  int s = bs & (SEQ - 1);
  int tok = seq[bs];
  x[idx] = emb[(size_t)tok * D_MODEL + d] + pes[(size_t)s * D_MODEL + d];
}

__global__ void addbn_kernel(float* __restrict__ x, const float* __restrict__ a,
                             const float* __restrict__ gamma,
                             const float* __restrict__ beta,
                             const float* __restrict__ mean,
                             const float* __restrict__ var, int n) {
  int idx = blockIdx.x * blockDim.x + threadIdx.x;
  if (idx >= n) return;
  int d = idx & 511;
  float v = x[idx] + a[idx];
  x[idx] = (v - mean[d]) * rsqrtf(var[d] + BN_EPS) * gamma[d] + beta[d];
}

// ---------------------------------------------------------------------------
// Driver. Input order = setup_inputs() dict with nested params flattened
// JAX-style (sorted keys; ASCII 'W' < 'b'):
//  0 sequence  1 encoder_output
//  params: 2 Wout | bn_bot: 3 beta 4 gamma 5 mean 6 var
//          bn_ffn: 7..10 | bn_mid: 11..14
//          bot: 15 Wk 16 Wo 17 Wq 18 Wv 19 bk 20 bo 21 bq 22 bv
//          23 bout 24 embed | ffn: 25 W1 26 W2 27 b1 28 b2
//          mid: 29 Wk 30 Wo 31 Wq 32 Wv 33 bk 34 bo 35 bq 36 bv | 37 pes
// ---------------------------------------------------------------------------
extern "C" void kernel_launch(void* const* d_in, const int* in_sizes, int n_in,
                              void* d_out, int out_size, void* d_ws,
                              size_t ws_size, hipStream_t stream) {
  (void)in_sizes; (void)n_in; (void)out_size; (void)ws_size;

  const int*   seq  = (const int*)d_in[0];
  const float* enc  = (const float*)d_in[1];
  const float* Wout = (const float*)d_in[2];
  const float* bnbot_beta = (const float*)d_in[3];
  const float* bnbot_gamma= (const float*)d_in[4];
  const float* bnbot_mean = (const float*)d_in[5];
  const float* bnbot_var  = (const float*)d_in[6];
  const float* bnffn_beta = (const float*)d_in[7];
  const float* bnffn_gamma= (const float*)d_in[8];
  const float* bnffn_mean = (const float*)d_in[9];
  const float* bnffn_var  = (const float*)d_in[10];
  const float* bnmid_beta = (const float*)d_in[11];
  const float* bnmid_gamma= (const float*)d_in[12];
  const float* bnmid_mean = (const float*)d_in[13];
  const float* bnmid_var  = (const float*)d_in[14];
  const float* bot_Wk = (const float*)d_in[15];
  const float* bot_Wo = (const float*)d_in[16];
  const float* bot_Wq = (const float*)d_in[17];
  const float* bot_Wv = (const float*)d_in[18];
  const float* bot_bk = (const float*)d_in[19];
  const float* bot_bo = (const float*)d_in[20];
  const float* bot_bq = (const float*)d_in[21];
  const float* bot_bv = (const float*)d_in[22];
  const float* bout   = (const float*)d_in[23];
  const float* emb    = (const float*)d_in[24];
  const float* ffn_W1 = (const float*)d_in[25];
  const float* ffn_W2 = (const float*)d_in[26];
  const float* ffn_b1 = (const float*)d_in[27];
  const float* ffn_b2 = (const float*)d_in[28];
  const float* mid_Wk = (const float*)d_in[29];
  const float* mid_Wo = (const float*)d_in[30];
  const float* mid_Wq = (const float*)d_in[31];
  const float* mid_Wv = (const float*)d_in[32];
  const float* mid_bk = (const float*)d_in[33];
  const float* mid_bo = (const float*)d_in[34];
  const float* mid_bq = (const float*)d_in[35];
  const float* mid_bv = (const float*)d_in[36];
  const float* pes    = (const float*)d_in[37];

  const int M = BATCH * SEQ;                 // 4096 rows (== BATCH*SEQ_ENC)
  const size_t T = (size_t)M * D_MODEL;      // 2M floats per activation tensor

  float* ws   = (float*)d_ws;
  float* x    = ws;
  float* q    = x + T;
  float* k    = q + T;
  float* v    = k + T;
  float* attn = v + T;
  float* a    = attn + T;
  float* hbuf = a + T;                       // M x DFF

  dim3 blk(128);
  auto gemm = [&](const float* A, const float* B, const float* bias, float* C,
                  int m, int n, int kk, int relu) {
    gemm_bf16_kernel<<<dim3(n / 64, m / 64), blk, 0, stream>>>(A, B, bias, C,
                                                               m, n, kk, relu);
  };

  {
    int n = (int)T;
    embed_kernel<<<(n + 255) / 256, 256, 0, stream>>>(seq, emb, pes, x, n);
  }

  for (int i = 0; i < NLAYER; ++i) {
    size_t wo = (size_t)i * D_MODEL * D_MODEL;
    size_t bo = (size_t)i * D_MODEL;
    size_t w1o = (size_t)i * D_MODEL * DFF;
    size_t b1o = (size_t)i * DFF;

    // --- causal self-attention ---
    gemm(x, bot_Wq + wo, bot_bq + bo, q, M, D_MODEL, D_MODEL, 0);
    gemm(x, bot_Wk + wo, bot_bk + bo, k, M, D_MODEL, D_MODEL, 0);
    gemm(x, bot_Wv + wo, bot_bv + bo, v, M, D_MODEL, D_MODEL, 0);
    attn_kernel<<<dim3(SEQ / 64, NHEAD, BATCH), blk, 0, stream>>>(
        q, k, v, attn, SEQ, SEQ, 1);
    gemm(attn, bot_Wo + wo, bot_bo + bo, a, M, D_MODEL, D_MODEL, 0);
    addbn_kernel<<<((int)T + 255) / 256, 256, 0, stream>>>(
        x, a, bnbot_gamma + bo, bnbot_beta + bo, bnbot_mean + bo,
        bnbot_var + bo, (int)T);

    // --- cross-attention over encoder output ---
    gemm(x, mid_Wq + wo, mid_bq + bo, q, M, D_MODEL, D_MODEL, 0);
    gemm(enc, mid_Wk + wo, mid_bk + bo, k, M, D_MODEL, D_MODEL, 0);
    gemm(enc, mid_Wv + wo, mid_bv + bo, v, M, D_MODEL, D_MODEL, 0);
    attn_kernel<<<dim3(SEQ / 64, NHEAD, BATCH), blk, 0, stream>>>(
        q, k, v, attn, SEQ, SEQ_ENC, 0);
    gemm(attn, mid_Wo + wo, mid_bo + bo, a, M, D_MODEL, D_MODEL, 0);
    addbn_kernel<<<((int)T + 255) / 256, 256, 0, stream>>>(
        x, a, bnmid_gamma + bo, bnmid_beta + bo, bnmid_mean + bo,
        bnmid_var + bo, (int)T);

    // --- FFN ---
    gemm(x, ffn_W1 + w1o, ffn_b1 + b1o, hbuf, M, DFF, D_MODEL, 1);
    gemm(hbuf, ffn_W2 + w1o, ffn_b2 + bo, a, M, D_MODEL, DFF, 0);
    addbn_kernel<<<((int)T + 255) / 256, 256, 0, stream>>>(
        x, a, bnffn_gamma + bo, bnffn_beta + bo, bnffn_mean + bo,
        bnffn_var + bo, (int)T);
  }

  // logits: [4096, 32000] — the bandwidth-dominant step
  gemm(x, Wout, bout, (float*)d_out, M, VOCAB, D_MODEL, 0);
}
